// GraphSeq2Seq_79328045957516
// MI455X (gfx1250) — matrix operand order repaired and checked by reading, not compiled, not run on hardware
//
#include <hip/hip_runtime.h>
#include <hip/hip_bf16.h>

// ---------------------------------------------------------------------------
// CDNA5 (gfx1250): bf16 WMMA everywhere, fragment-order packed operands so the
// inner loops are pure b128-load + v_wmma_f32_16x16x32_bf16 streams.
// ---------------------------------------------------------------------------

typedef __attribute__((ext_vector_type(16))) __bf16          v16bf;
typedef __attribute__((ext_vector_type(16))) unsigned short  v16us;
typedef __attribute__((ext_vector_type(8)))  float           v8f;

// ---- bf16 helpers ---------------------------------------------------------
__device__ __forceinline__ unsigned short f2bf(float f) {
  unsigned int u = __float_as_uint(f);
  u += 0x7fffu + ((u >> 16) & 1u);
  return (unsigned short)(u >> 16);
}
__device__ __forceinline__ float sigm(float x) { return 1.f / (1.f + __expf(-x)); }

// ---- fragment-order packing (inverse of CDNA5 wave32 WMMA layouts) --------
// A (16x32 bf16): lane0-15 row=lane, K 0..7 (v0-3) / 16..23 (v4-7);
//                 lane16-31 same rows, K 8..15 / 24..31.
__device__ __forceinline__ int apack_off(int m, int k, int ktiles) {
  int tile = (m >> 4) * ktiles + (k >> 5);
  int kk = k & 31, q = kk >> 3;
  int lane = (m & 15) + ((q & 1) << 4);
  int e = ((((q >> 1) << 2) + ((kk & 7) >> 1)) << 1) + (kk & 1);
  return tile * 512 + lane * 16 + e;
}
// B (32x16 bf16): lane = col + 16*((k>>3)&1), e = 8*(k>=16 within tile) + (k&7).
__device__ __forceinline__ int bpack_off(int k, int n, int ktiles) {
  int tile = (n >> 4) * ktiles + (k >> 5);
  int kk = k & 31;
  int lane = (n & 15) + (((kk >> 3) & 1) << 4);
  int e = ((kk >> 4) << 3) + (kk & 7);
  return tile * 512 + lane * 16 + e;
}
// One packed fragment = 32 contiguous bytes per lane.
__device__ __forceinline__ v16bf load_frag_pk(const unsigned short* P, int tile,
                                              int lane) {
  union { v16us u; v16bf b; } fc;
  fc.u = *(const v16us*)(P + (size_t)tile * 512 + lane * 16);
  return fc.b;
}
__device__ __forceinline__ v8f wmma_bf16(v16bf a, v16bf b, v8f c) {
  return __builtin_amdgcn_wmma_f32_16x16x32_bf16(false, a, false, b, (short)0, c,
                                                 false, false);
}

// ---------------------------------------------------------------------------
// graph_norm: x viewed [B, F, 512]; per (b,g) mean/var over 512 elements.
// ---------------------------------------------------------------------------
__global__ void __launch_bounds__(128) graph_norm_k(const float* x, const float* sc,
                                                    const float* sh, const float* al,
                                                    float* y, int F) {
  __shared__ float s1[128], s2[128];
  int g = blockIdx.x, b = blockIdx.y, tid = threadIdx.x;
  const float* p = x + ((size_t)b * F + g) * 512;
  float a = 0.f, a2 = 0.f;
  for (int j = tid; j < 512; j += 128) { float v = p[j]; a += v; a2 += v * v; }
  s1[tid] = a; s2[tid] = a2; __syncthreads();
  for (int off = 64; off; off >>= 1) {
    if (tid < off) { s1[tid] += s1[tid + off]; s2[tid] += s2[tid + off]; }
    __syncthreads();
  }
  float mean = s1[0] * (1.f / 512.f);
  float var  = s2[0] * (1.f / 512.f) - mean * mean;
  float inv  = rsqrtf(var + 1e-5f);
  float scv = sc[g], shv = sh[g], alv = al[g];
  float* q = y + ((size_t)b * F + g) * 512;
  for (int j = tid; j < 512; j += 128) q[j] = (p[j] - alv * mean) * inv * scv + shv;
}

// ---------------------------------------------------------------------------
// pack kernels: f32 -> fragment-ordered bf16
// ---------------------------------------------------------------------------
__global__ void pack_a_k(const float* A, unsigned short* Ap, int M, int K) {
  int i = blockIdx.x * 256 + threadIdx.x;
  if (i >= M * K) return;
  int m = i / K, k = i - m * K;
  Ap[apack_off(m, k, K >> 5)] = f2bf(A[i]);
}
template <int TRANS>
__global__ void pack_b_k(const float* W, unsigned short* Wp, int K, int N) {
  int i = blockIdx.x * 256 + threadIdx.x;
  if (i >= K * N) return;
  int k = i / N, n = i - k * N;
  float v = TRANS ? W[(size_t)n * K + k] : W[i];
  Wp[bpack_off(k, n, K >> 5)] = f2bf(v);
}
// per-(b,h) slab [512 x nh] B-pack of Wh_r for GAT aggregation
__global__ void pack_whr_k(const float* Wh, unsigned short* Wp, int total, int nh) {
  int i = blockIdx.x * 256 + threadIdx.x;
  if (i >= total) return;
  int slab = 512 * nh;
  int bh = i / slab, r = i - bh * slab;
  int j = r / nh, c = r - j * nh;
  Wp[(size_t)bh * slab + bpack_off(j, c, 16)] = f2bf(Wh[i]);
}
__global__ void actions_f32_k(const int* a, float* o, int n) {
  int i = blockIdx.x * 256 + threadIdx.x;
  if (i < n) o[i] = (float)a[i];
}

// ---------------------------------------------------------------------------
// Packed GEMM: C[MxN] = act(Ap @ Wp); one wave per 16x16 tile, K fully unrolled.
// ACT: 0=none, 1=leaky(0.2)
// ---------------------------------------------------------------------------
template <int K, int N, int ACT>
__global__ void __launch_bounds__(32) gemm_pk(const unsigned short* __restrict__ Ap,
                                              const unsigned short* __restrict__ Wp,
                                              float* __restrict__ C) {
  constexpr int KT = K / 32;
  int lane = threadIdx.x & 31;
  int nt = blockIdx.x, mt = blockIdx.y;
  v8f acc;
#pragma unroll
  for (int v = 0; v < 8; ++v) acc[v] = 0.f;
#pragma unroll
  for (int kt = 0; kt < KT; ++kt) {
    v16bf a = load_frag_pk(Ap, mt * KT + kt, lane);
    v16bf b = load_frag_pk(Wp, nt * KT + kt, lane);
    acc = wmma_bf16(a, b, acc);
  }
#pragma unroll
  for (int v = 0; v < 8; ++v) {
    int m = mt * 16 + v + ((lane & 16) ? 8 : 0), n = nt * 16 + (lane & 15);
    float xv = acc[v];
    if (ACT == 1) xv = xv > 0.f ? xv : 0.2f * xv;
    C[(size_t)m * N + n] = xv;
  }
}

// ---------------------------------------------------------------------------
// el/er head dots
// ---------------------------------------------------------------------------
__global__ void headdot_k(const float* Wh, const float* a, float* out, int n, int nh) {
  int i = blockIdx.x * 256 + threadIdx.x;
  if (i >= n) return;
  const float* p = Wh + (size_t)i * nh;
  float s = 0.f;
  for (int k = 0; k < nh; ++k) s += p[k] * a[k];
  out[i] = s;
}

// ---------------------------------------------------------------------------
// Fused GAT aggregation: masked e, row softmax over 512, att (packed bf16 in
// LDS) @ Wh_r (packed bf16) via WMMA, elu.  grid=(32 row tiles, B*H), block 256.
// ---------------------------------------------------------------------------
__global__ void __launch_bounds__(256) gat_agg_k(const float* el, const float* er,
                                                 const float* adj,
                                                 const unsigned short* __restrict__ Whpk,
                                                 float* out, int H, int nh) {
  __shared__ float e_s[16][512];
  __shared__ alignas(32) unsigned short att_pk[16 * 512];
  __shared__ float red[16][16];
  int tid = threadIdx.x;
  int bh = blockIdx.y, b = bh / H, m0 = blockIdx.x * 16;
  const float* elp  = el + (size_t)bh * 512;
  const float* erp  = er + (size_t)bh * 512;
  const float* adjp = adj + (size_t)b * 512 * 512;
  for (int i = tid; i < 16 * 512; i += 256) {
    int r = i >> 9, j = i & 511;
    float ev = elp[m0 + r] + erp[j];
    e_s[r][j] = (adjp[(size_t)(m0 + r) * 512 + j] > 0.f) ? ev : -9.0e15f;
  }
  __syncthreads();
  int r = tid >> 4, l = tid & 15;
  float mx = -3.0e38f;
  for (int j = l; j < 512; j += 16) mx = fmaxf(mx, e_s[r][j]);
  red[r][l] = mx; __syncthreads();
  if (l == 0) { float m2 = red[r][0]; for (int i = 1; i < 16; ++i) m2 = fmaxf(m2, red[r][i]); red[r][0] = m2; }
  __syncthreads();
  mx = red[r][0];
  float s = 0.f;
  for (int j = l; j < 512; j += 16) { float e = __expf(e_s[r][j] - mx); e_s[r][j] = e; s += e; }
  __syncthreads();
  red[r][l] = s; __syncthreads();
  if (l == 0) { float a = 0.f; for (int i = 0; i < 16; ++i) a += red[r][i]; red[r][0] = a; }
  __syncthreads();
  float inv = 1.f / red[r][0];
  for (int j = l; j < 512; j += 16) att_pk[apack_off(r, j, 16)] = f2bf(e_s[r][j] * inv);
  __syncthreads();

  int lane = tid & 31, wave = tid >> 5;
  const unsigned short* Wb = Whpk + (size_t)bh * 512 * nh;
  float* ob = out + (size_t)bh * 512 * nh;
  for (int nt = wave; nt < nh / 16; nt += 8) {
    v8f acc;
#pragma unroll
    for (int v = 0; v < 8; ++v) acc[v] = 0.f;
#pragma unroll
    for (int kt = 0; kt < 16; ++kt) {
      v16bf a = load_frag_pk(att_pk, kt, lane);
      v16bf bb = load_frag_pk(Wb, nt * 16 + kt, lane);
      acc = wmma_bf16(a, bb, acc);
    }
#pragma unroll
    for (int v = 0; v < 8; ++v) {
      int m = v + ((lane & 16) ? 8 : 0), n = lane & 15;
      float xv = acc[v];
      xv = xv > 0.f ? xv : (__expf(xv) - 1.f);  // elu
      ob[(size_t)(m0 + m) * nh + nt * 16 + n] = xv;
    }
  }
}

// ---------------------------------------------------------------------------
// Workgroup packed GEMM (persistent kernels): C[32xNN] = A1@W1 (+ A2@W2).
// ---------------------------------------------------------------------------
template <int K1, int K2, int NN>
__device__ void wg_gemm_pk(const unsigned short* __restrict__ A1,
                           const unsigned short* __restrict__ W1,
                           const unsigned short* __restrict__ A2,
                           const unsigned short* __restrict__ W2,
                           float* __restrict__ Cg, int tid) {
  constexpr int KT1 = K1 / 32, KT2 = K2 / 32, NTILES = (NN / 16) * 2;
  int lane = tid & 31, wave = tid >> 5;
  for (int tt = wave; tt < NTILES; tt += 32) {
    int mt = tt & 1, nt = tt >> 1;
    v8f acc;
#pragma unroll
    for (int v = 0; v < 8; ++v) acc[v] = 0.f;
#pragma unroll
    for (int kt = 0; kt < KT1; ++kt) {
      __builtin_prefetch(W1 + (size_t)(nt * KT1 + kt + 1) * 512 + lane * 16, 0, 0);
      v16bf a = load_frag_pk(A1, mt * KT1 + kt, lane);
      v16bf b = load_frag_pk(W1, nt * KT1 + kt, lane);
      acc = wmma_bf16(a, b, acc);
    }
    if constexpr (K2 > 0) {
#pragma unroll
      for (int kt = 0; kt < KT2; ++kt) {
        v16bf a = load_frag_pk(A2, mt * KT2 + kt, lane);
        v16bf b = load_frag_pk(W2, nt * KT2 + kt, lane);
        acc = wmma_bf16(a, b, acc);
      }
    }
#pragma unroll
    for (int v = 0; v < 8; ++v) {
      int m = mt * 16 + v + ((lane & 16) ? 8 : 0), n = nt * 16 + (lane & 15);
      Cg[(size_t)m * NN + n] = acc[v];
    }
  }
}

// LSTM cell: gates[32][1024] (+bias) -> h,c; h also stored packed bf16 (K=256).
__device__ void lstm_cell_dev(const float* gates, const float* bias, float* c,
                              float* h, unsigned short* hpk, float* out_opt,
                              size_t out_bstride, int tid) {
  for (int i = tid; i < 32 * 256; i += 1024) {
    int b = i >> 8, u = i & 255;
    const float* gp = gates + (size_t)b * 1024;
    float gi = gp[u] + bias[u];
    float gf = gp[256 + u] + bias[256 + u];
    float gg = gp[512 + u] + bias[512 + u];
    float go = gp[768 + u] + bias[768 + u];
    float cc = sigm(gf) * c[i] + sigm(gi) * tanhf(gg);
    float hh = sigm(go) * tanhf(cc);
    c[i] = cc; h[i] = hh;
    hpk[apack_off(b, u, 8)] = f2bf(hh);
    if (out_opt) out_opt[(size_t)b * out_bstride + u] = hh;
  }
}

// ---------------------------------------------------------------------------
// Persistent encoder LSTM: one WG (32 wave32 waves), 512 timesteps.
// ---------------------------------------------------------------------------
__global__ void __launch_bounds__(1024) enc_lstm_k(
    const float* xseq, const unsigned short* Wih0t, const unsigned short* Whh0t,
    const unsigned short* Wih1t, const unsigned short* Whh1t,
    const float* b0, const float* b1, float* h_state, float* c_state,
    unsigned short* hpk0, unsigned short* hpk1, float* gates, float* enc_out) {
  __shared__ alignas(32) unsigned short xpk[32 * 256];
  int tid = threadIdx.x;
  for (int i = tid; i < 8192; i += 1024) {
    h_state[i] = 0.f; c_state[i] = 0.f;
    h_state[8192 + i] = 0.f; c_state[8192 + i] = 0.f;
    hpk0[i] = 0; hpk1[i] = 0;
  }
  __threadfence(); __syncthreads();
  for (int t = 0; t < 512; ++t) {
    for (int i = tid; i < 8192; i += 1024) {
      int b = i >> 8, u = i & 255;
      xpk[apack_off(b, u, 8)] = f2bf(xseq[((size_t)b * 512 + t) * 256 + u]);
    }
    __syncthreads();
    wg_gemm_pk<256, 256, 1024>(xpk, Wih0t, hpk0, Whh0t, gates, tid);
    __threadfence(); __syncthreads();
    lstm_cell_dev(gates, b0, c_state, h_state, hpk0, nullptr, 0, tid);
    __threadfence(); __syncthreads();
    wg_gemm_pk<256, 256, 1024>(hpk0, Wih1t, hpk1, Whh1t, gates, tid);
    __threadfence(); __syncthreads();
    lstm_cell_dev(gates, b1, c_state + 8192, h_state + 8192, hpk1,
                  enc_out + (size_t)t * 256, (size_t)512 * 256, tid);
    __threadfence(); __syncthreads();
  }
}

// ---------------------------------------------------------------------------
// Persistent decoder: emb-LN -> 2-layer LSTM -> LN -> Luong attention ->
// concat proj -> tanh ctx -> actor softmax + critic, 512 steps.
// ---------------------------------------------------------------------------
__global__ void __launch_bounds__(1024) dec_lstm_k(
    const int* actions, const float* emb, const float* emb_g, const float* emb_b,
    const unsigned short* Wih0t, const unsigned short* Whh0t,
    const unsigned short* Wih1t, const unsigned short* Whh1t,
    const float* b0, const float* b1, const float* ln_g, const float* ln_b,
    const float* actor_W, const float* actor_b, const float* critic_W,
    const float* critic_b, const unsigned short* catWt, const float* enc_out,
    const float* enc_proj, float* h_state, float* c_state, unsigned short* hpk0,
    unsigned short* hpk1, float* gates, float* ctx, float* outln, float* catout,
    float* out_logits, float* out_values) {
  __shared__ alignas(32) unsigned short xpk[32 * 512];
  __shared__ float redA[32][32], redB[32][32];
  __shared__ int decin[32];
  __shared__ float pib[32][2];
  int tid = threadIdx.x;
  for (int i = tid; i < 8192; i += 1024) {
    int b = i >> 8, u = i & 255;
    const float* p = enc_out + (size_t)b * 512 * 256 + u;
    float s = 0.f;
    for (int j = 0; j < 512; ++j) s += p[(size_t)j * 256];
    ctx[i] = s * (1.f / 512.f);
    hpk0[apack_off(b, u, 8)] = f2bf(h_state[i]);
    hpk1[apack_off(b, u, 8)] = f2bf(h_state[8192 + i]);
  }
  __threadfence(); __syncthreads();

  for (int t = 0; t < 512; ++t) {
    if (tid < 32) decin[tid] = (t == 0) ? 0 : actions[tid * 512 + (t - 1)];
    __syncthreads();
    int b = tid >> 5, l = tid & 31;
    // embedding layer-norm -> xpk K cols 0:256
    {
      const float* e = emb + (size_t)decin[b] * 256;
      float s = 0.f, s2 = 0.f;
      for (int u = l; u < 256; u += 32) { float v = e[u]; s += v; s2 += v * v; }
      redA[b][l] = s; redB[b][l] = s2; __syncthreads();
      if (l == 0) {
        float a = 0.f, a2 = 0.f;
        for (int i = 0; i < 32; ++i) { a += redA[b][i]; a2 += redB[b][i]; }
        float mean = a * (1.f / 256.f), var = a2 * (1.f / 256.f) - mean * mean;
        redA[b][0] = mean; redB[b][0] = rsqrtf(var + 1e-5f);
      }
      __syncthreads();
      float mean = redA[b][0], inv = redB[b][0];
      for (int u = l; u < 256; u += 32)
        xpk[apack_off(b, u, 16)] = f2bf((e[u] - mean) * inv * emb_g[u] + emb_b[u]);
    }
    for (int i = tid; i < 8192; i += 1024) {
      int bb = i >> 8, u = i & 255;
      xpk[apack_off(bb, 256 + u, 16)] = f2bf(ctx[i]);
    }
    __syncthreads();
    wg_gemm_pk<512, 256, 1024>(xpk, Wih0t, hpk0, Whh0t, gates, tid);
    __threadfence(); __syncthreads();
    lstm_cell_dev(gates, b0, c_state, h_state, hpk0, nullptr, 0, tid);
    __threadfence(); __syncthreads();
    wg_gemm_pk<256, 256, 1024>(hpk0, Wih1t, hpk1, Whh1t, gates, tid);
    __threadfence(); __syncthreads();
    lstm_cell_dev(gates, b1, c_state + 8192, h_state + 8192, hpk1, nullptr, 0, tid);
    __threadfence(); __syncthreads();
    // output layer-norm -> outln (f32) and xpk cols 0:256
    {
      const float* hrow = h_state + 8192 + (size_t)b * 256;
      float s = 0.f, s2 = 0.f;
      for (int u = l; u < 256; u += 32) { float v = hrow[u]; s += v; s2 += v * v; }
      redA[b][l] = s; redB[b][l] = s2; __syncthreads();
      if (l == 0) {
        float a = 0.f, a2 = 0.f;
        for (int i = 0; i < 32; ++i) { a += redA[b][i]; a2 += redB[b][i]; }
        float mean = a * (1.f / 256.f), var = a2 * (1.f / 256.f) - mean * mean;
        redA[b][0] = mean; redB[b][0] = rsqrtf(var + 1e-5f);
      }
      __syncthreads();
      float mean = redA[b][0], inv = redB[b][0];
      for (int u = l; u < 256; u += 32) {
        float v = (hrow[u] - mean) * inv * ln_g[u] + ln_b[u];
        outln[(size_t)b * 256 + u] = v;
        xpk[apack_off(b, u, 16)] = f2bf(v);
      }
    }
    __threadfence(); __syncthreads();
    // score[b][j] (stored in gates scratch)
    for (int i = tid; i < 16384; i += 1024) {
      int bb = i >> 9, j = i & 511;
      const float* o = outln + (size_t)bb * 256;
      const float* ep = enc_proj + ((size_t)bb * 512 + j) * 256;
      float s = 0.f;
      for (int u = 0; u < 256; ++u) s += o[u] * ep[u];
      gates[(size_t)bb * 512 + j] = s;
    }
    __threadfence(); __syncthreads();
    // softmax over j
    {
      float* sc = gates + (size_t)b * 512;
      float mx = -3.0e38f;
      for (int j = l; j < 512; j += 32) mx = fmaxf(mx, sc[j]);
      redA[b][l] = mx; __syncthreads();
      if (l == 0) { float m2 = redA[b][0]; for (int i = 1; i < 32; ++i) m2 = fmaxf(m2, redA[b][i]); redA[b][0] = m2; }
      __syncthreads();
      mx = redA[b][0];
      float s = 0.f;
      for (int j = l; j < 512; j += 32) { float e = __expf(sc[j] - mx); sc[j] = e; s += e; }
      __syncthreads();
      redB[b][l] = s; __syncthreads();
      if (l == 0) { float a = 0.f; for (int i = 0; i < 32; ++i) a += redB[b][i]; redB[b][0] = 1.f / a; }
      __syncthreads();
      float inv = redB[b][0];
      for (int j = l; j < 512; j += 32) sc[j] *= inv;
    }
    __threadfence(); __syncthreads();
    // attn_ctx -> xpk cols 256:512
    for (int i = tid; i < 8192; i += 1024) {
      int bb = i >> 8, u = i & 255;
      const float* w = gates + (size_t)bb * 512;
      const float* p = enc_out + (size_t)bb * 512 * 256 + u;
      float s = 0.f;
      for (int j = 0; j < 512; ++j) s += w[j] * p[(size_t)j * 256];
      xpk[apack_off(bb, 256 + u, 16)] = f2bf(s);
    }
    __syncthreads();
    wg_gemm_pk<512, 0, 256>(xpk, catWt, nullptr, nullptr, catout, tid);
    __threadfence(); __syncthreads();
    for (int i = tid; i < 8192; i += 1024) ctx[i] = tanhf(catout[i]);
    __threadfence(); __syncthreads();
    if (tid < 64) {
      int bb = tid >> 1, o = tid & 1;
      const float* c0 = ctx + (size_t)bb * 256;
      const float* w = actor_W + (size_t)o * 256;
      float s = actor_b[o];
      for (int u = 0; u < 256; ++u) s += c0[u] * w[u];
      pib[bb][o] = s;
    } else if (tid < 96) {
      int bb = tid - 64;
      const float* c0 = ctx + (size_t)bb * 256;
      float s = critic_b[0];
      for (int u = 0; u < 256; ++u) s += c0[u] * critic_W[u];
      out_values[(size_t)bb * 512 + t] = s;
    }
    __syncthreads();
    if (tid < 32) {
      int bb = tid;
      float m = fmaxf(pib[bb][0], pib[bb][1]);
      float e0 = __expf(pib[bb][0] - m), e1 = __expf(pib[bb][1] - m);
      float inv = 1.f / (e0 + e1);
      out_logits[((size_t)bb * 512 + t) * 2 + 0] = e0 * inv;
      out_logits[((size_t)bb * 512 + t) * 2 + 1] = e1 * inv;
    }
    __threadfence(); __syncthreads();
  }
}

// ---------------------------------------------------------------------------
extern "C" void kernel_launch(void* const* d_in, const int* in_sizes, int n_in,
                              void* d_out, int out_size, void* d_ws, size_t ws_size,
                              hipStream_t stream) {
  const float* x        = (const float*)d_in[0];
  const float* adj      = (const float*)d_in[1];
  const int*   actions  = (const int*)d_in[2];
  const float* gn1_s = (const float*)d_in[3];
  const float* gn1_h = (const float*)d_in[4];
  const float* gn1_a = (const float*)d_in[5];
  const float* W_l1  = (const float*)d_in[6];
  const float* W_r1  = (const float*)d_in[7];
  const float* a1    = (const float*)d_in[8];
  const float* gn2_s = (const float*)d_in[9];
  const float* gn2_h = (const float*)d_in[10];
  const float* gn2_a = (const float*)d_in[11];
  const float* W_l2  = (const float*)d_in[12];
  const float* W_r2  = (const float*)d_in[13];
  const float* a2    = (const float*)d_in[14];
  const float* e_wih0 = (const float*)d_in[15];
  const float* e_whh0 = (const float*)d_in[16];
  const float* e_b0   = (const float*)d_in[17];
  const float* e_wih1 = (const float*)d_in[18];
  const float* e_whh1 = (const float*)d_in[19];
  const float* e_b1   = (const float*)d_in[20];
  const float* emb    = (const float*)d_in[21];
  const float* emb_g  = (const float*)d_in[22];
  const float* emb_b  = (const float*)d_in[23];
  const float* d_wih0 = (const float*)d_in[24];
  const float* d_whh0 = (const float*)d_in[25];
  const float* d_b0   = (const float*)d_in[26];
  const float* d_wih1 = (const float*)d_in[27];
  const float* d_whh1 = (const float*)d_in[28];
  const float* d_b1   = (const float*)d_in[29];
  const float* ln_g   = (const float*)d_in[30];
  const float* ln_b   = (const float*)d_in[31];
  const float* attn_W = (const float*)d_in[32];
  const float* concat_W = (const float*)d_in[33];
  const float* actor_W  = (const float*)d_in[34];
  const float* actor_b  = (const float*)d_in[35];
  const float* critic_W = (const float*)d_in[36];
  const float* critic_b = (const float*)d_in[37];

  char* wsb = (char*)d_ws;
  size_t off = 0;
  auto allocF = [&](size_t n) { float* p = (float*)(wsb + off);
                                off = (off + n * 4 + 255) & ~(size_t)255; return p; };
  auto allocU = [&](size_t n) { unsigned short* p = (unsigned short*)(wsb + off);
                                off = (off + n * 2 + 255) & ~(size_t)255; return p; };

  float* h0     = allocF(1048576);
  float* whl1   = allocF(2097152);
  float* whr1   = allocF(2097152);
  float* el1    = allocF(131072);
  float* er1    = allocF(131072);
  float* h1     = allocF(2097152);
  float* h1n    = allocF(2097152);
  float* whl2   = allocF(4194304);
  float* whr2   = allocF(4194304);
  float* el2    = allocF(16384);
  float* er2    = allocF(16384);
  float* h2     = allocF(4194304);
  float* encout = allocF(4194304);
  float* encprj = allocF(4194304);
  float* hst    = allocF(16384);
  float* cst    = allocF(16384);
  float* gates  = allocF(32768);
  float* ctx    = allocF(8192);
  float* outln  = allocF(8192);
  float* catout = allocF(8192);
  unsigned short* h0pk    = allocU(1048576);   // packed A [16384 x 64]
  unsigned short* h1npk   = allocU(2097152);   // packed A [16384 x 128]
  unsigned short* h2pk    = allocU(4194304);   // packed A [16384 x 256] (enc_out)
  unsigned short* Wl1p    = allocU(8192);
  unsigned short* Wr1p    = allocU(8192);
  unsigned short* Wl2p    = allocU(32768);
  unsigned short* Wr2p    = allocU(32768);
  unsigned short* attnWp  = allocU(65536);
  unsigned short* whr1pk  = allocU(2097152);
  unsigned short* whr2pk  = allocU(4194304);
  unsigned short* eWih0t  = allocU(262144);
  unsigned short* eWhh0t  = allocU(262144);
  unsigned short* eWih1t  = allocU(262144);
  unsigned short* eWhh1t  = allocU(262144);
  unsigned short* dWih0t  = allocU(524288);
  unsigned short* dWhh0t  = allocU(262144);
  unsigned short* dWih1t  = allocU(262144);
  unsigned short* dWhh1t  = allocU(262144);
  unsigned short* catWt   = allocU(131072);
  unsigned short* hpk0    = allocU(8192);
  unsigned short* hpk1    = allocU(8192);
  (void)ws_size; (void)n_in; (void)in_sizes; (void)out_size;

  float* outF   = (float*)d_out;
  float* outAct = outF;
  float* outLog = outF + 16384;
  float* outVal = outF + 49152;

  // ---- GAT layer 1 ----
  graph_norm_k<<<dim3(64, 32), 128, 0, stream>>>(x, gn1_s, gn1_h, gn1_a, h0, 64);
  pack_a_k<<<4096, 256, 0, stream>>>(h0, h0pk, 16384, 64);
  pack_b_k<0><<<32, 256, 0, stream>>>(W_l1, Wl1p, 64, 128);
  pack_b_k<0><<<32, 256, 0, stream>>>(W_r1, Wr1p, 64, 128);
  gemm_pk<64, 128, 1><<<dim3(8, 1024), 32, 0, stream>>>(h0pk, Wl1p, whl1);
  gemm_pk<64, 128, 1><<<dim3(8, 1024), 32, 0, stream>>>(h0pk, Wr1p, whr1);
  pack_whr_k<<<8192, 256, 0, stream>>>(whr1, whr1pk, 2097152, 16);
  headdot_k<<<512, 256, 0, stream>>>(whl1, a1, el1, 131072, 16);
  headdot_k<<<512, 256, 0, stream>>>(whr1, a1, er1, 131072, 16);
  gat_agg_k<<<dim3(32, 256), 256, 0, stream>>>(el1, er1, adj, whr1pk, h1, 8, 16);

  // ---- GAT layer 2 ----
  graph_norm_k<<<dim3(128, 32), 128, 0, stream>>>(h1, gn2_s, gn2_h, gn2_a, h1n, 128);
  pack_a_k<<<8192, 256, 0, stream>>>(h1n, h1npk, 16384, 128);
  pack_b_k<0><<<128, 256, 0, stream>>>(W_l2, Wl2p, 128, 256);
  pack_b_k<0><<<128, 256, 0, stream>>>(W_r2, Wr2p, 128, 256);
  gemm_pk<128, 256, 1><<<dim3(16, 1024), 32, 0, stream>>>(h1npk, Wl2p, whl2);
  gemm_pk<128, 256, 1><<<dim3(16, 1024), 32, 0, stream>>>(h1npk, Wr2p, whr2);
  pack_whr_k<<<16384, 256, 0, stream>>>(whr2, whr2pk, 4194304, 256);
  headdot_k<<<64, 256, 0, stream>>>(whl2, a2, el2, 16384, 256);
  headdot_k<<<64, 256, 0, stream>>>(whr2, a2, er2, 16384, 256);
  gat_agg_k<<<dim3(32, 32), 256, 0, stream>>>(el2, er2, adj, whr2pk, h2, 1, 256);

  // ---- LSTM weight packing ----
  pack_b_k<1><<<1024, 256, 0, stream>>>(e_wih0, eWih0t, 256, 1024);
  pack_b_k<1><<<1024, 256, 0, stream>>>(e_whh0, eWhh0t, 256, 1024);
  pack_b_k<1><<<1024, 256, 0, stream>>>(e_wih1, eWih1t, 256, 1024);
  pack_b_k<1><<<1024, 256, 0, stream>>>(e_whh1, eWhh1t, 256, 1024);
  pack_b_k<1><<<2048, 256, 0, stream>>>(d_wih0, dWih0t, 512, 1024);
  pack_b_k<1><<<1024, 256, 0, stream>>>(d_whh0, dWhh0t, 256, 1024);
  pack_b_k<1><<<1024, 256, 0, stream>>>(d_wih1, dWih1t, 256, 1024);
  pack_b_k<1><<<1024, 256, 0, stream>>>(d_whh1, dWhh1t, 256, 1024);
  pack_b_k<1><<<512, 256, 0, stream>>>(concat_W, catWt, 512, 256);
  pack_b_k<1><<<256, 256, 0, stream>>>(attn_W, attnWp, 256, 256);

  // ---- Encoder (persistent) ----
  enc_lstm_k<<<1, 1024, 0, stream>>>(h2, eWih0t, eWhh0t, eWih1t, eWhh1t, e_b0, e_b1,
                                     hst, cst, hpk0, hpk1, gates, encout);
  // enc_proj = enc_out @ attn_W^T
  pack_a_k<<<16384, 256, 0, stream>>>(encout, h2pk, 16384, 256);
  gemm_pk<256, 256, 0><<<dim3(16, 1024), 32, 0, stream>>>(h2pk, attnWp, encprj);
  // ---- Decoder (persistent) ----
  dec_lstm_k<<<1, 1024, 0, stream>>>(actions, emb, emb_g, emb_b, dWih0t, dWhh0t,
                                     dWih1t, dWhh1t, d_b0, d_b1, ln_g, ln_b,
                                     actor_W, actor_b, critic_W, critic_b, catWt,
                                     encout, encprj, hst, cst, hpk0, hpk1, gates,
                                     ctx, outln, catout, outLog, outVal);
  actions_f32_k<<<64, 256, 0, stream>>>(actions, outAct, 16384);
}
